// GroupedExperts_3659312136193
// MI455X (gfx1250) — matrix operand order repaired and checked by reading, not compiled, not run on hardware
//
#include <hip/hip_runtime.h>
#include <cstddef>

// Problem constants (match reference setup_inputs)
#define NE   8        // experts
#define DM   512      // d_model
#define HD   1408     // hidden
#define TT   32768    // total tokens
#define TPE  4096     // tokens per expert

typedef __attribute__((ext_vector_type(16))) __bf16          v16bf;
typedef __attribute__((ext_vector_type(8)))  float           v8f;
typedef __attribute__((ext_vector_type(8)))  unsigned short  v8u;
typedef __attribute__((ext_vector_type(16))) unsigned short  v16u;

// round-to-nearest-even f32 -> bf16 (bits)
__device__ __forceinline__ unsigned short f2bf(float f) {
  unsigned u = __builtin_bit_cast(unsigned, f);
  u += 0x7FFFu + ((u >> 16) & 1u);
  return (unsigned short)(u >> 16);
}

#define LDSS 72   // padded LDS row stride in u16 (144 B): lanes tile all 64 banks -> conflict-free b128

// A-fragment (16x32 bf16, row-major tile in LDS), CDNA5 16-bit A layout:
// lanes 0-15: row=lane, K0..7 (bytes 0..15) + K16..23 (bytes 32..47)
// lanes 16-31: row=lane-16, K8..15 (bytes 16..31) + K24..31 (bytes 48..63)
__device__ __forceinline__ v16bf load_frag_a(const unsigned short* t, int lane, int kk) {
  const int r = lane & 15, hh = lane >> 4;
  const unsigned short* p = t + r * LDSS + kk * 32 + hh * 8;
  v8u lo = *(const v8u*)(p);
  v8u hi = *(const v8u*)(p + 16);
  v16u v = __builtin_shufflevector(lo, hi, 0,1,2,3,4,5,6,7,8,9,10,11,12,13,14,15);
  return __builtin_bit_cast(v16bf, v);
}

// B-fragment from row-major (N x K) tile (w12 / w3 native layout):
// lane = column n (lane&15); lanes 0-15 hold K0..7 + K16..23, lanes 16-31 hold K8..15 + K24..31
__device__ __forceinline__ v16bf load_frag_b(const unsigned short* t, int lane, int kk) {
  const int n = lane & 15, hh = lane >> 4;
  const unsigned short* p = t + n * LDSS + kk * 32 + hh * 8;
  v8u lo = *(const v8u*)(p);
  v8u hi = *(const v8u*)(p + 16);
  v16u v = __builtin_shufflevector(lo, hi, 0,1,2,3,4,5,6,7,8,9,10,11,12,13,14,15);
  return __builtin_bit_cast(v16bf, v);
}

__device__ __forceinline__ v8f wmma_bf16(v16bf a, v16bf b, v8f c) {
  return __builtin_amdgcn_wmma_f32_16x16x32_bf16(false, a, false, b, (short)0, c, false, false);
}

// ---------------- elementwise f32 -> bf16 ----------------
__global__ __launch_bounds__(256)
void cvt_f32_bf16(const float* __restrict__ src, unsigned short* __restrict__ dst, int n4) {
  int i = blockIdx.x * blockDim.x + threadIdx.x;
  const int stride = gridDim.x * blockDim.x;
  for (; i < n4; i += stride) {
    float4 f = ((const float4*)src)[i];
    unsigned long long pk =
        (unsigned long long)f2bf(f.x) |
        ((unsigned long long)f2bf(f.y) << 16) |
        ((unsigned long long)f2bf(f.z) << 32) |
        ((unsigned long long)f2bf(f.w) << 48);
    ((unsigned long long*)dst)[i] = pk;
  }
}

// ---------------- GEMM1 + SwiGLU ----------------
// block = 256 threads (8 waves), tile = 128 tokens x 128 j-columns (gate AND up = 256 B rows)
// waves: 2 (token, 64 rows) x 4 (j, 32 cols); each wave: 4x2 16x16 tiles per matrix
// grid.x = TT/128 token tiles, grid.y = HD/128 j tiles
__global__ __launch_bounds__(256)
void gemm1_swiglu(const unsigned short* __restrict__ xb,
                  const unsigned short* __restrict__ w12b,
                  unsigned short* __restrict__ hb) {
  __shared__ __align__(16) unsigned short ldsA[128 * LDSS];
  __shared__ __align__(16) unsigned short ldsB[256 * LDSS];  // rows 0-127 gate, 128-255 up

  const int tid  = threadIdx.x;
  const int lane = tid & 31;
  const int wave = tid >> 5;
  const int wm   = wave & 1;    // token sub-tile (64 rows)
  const int wn   = wave >> 1;   // j sub-tile (32 cols)
  const int tok0 = blockIdx.x * 128;
  const int e    = blockIdx.x >> 5;   // 4096/128 = 32 tiles per expert
  const int j0   = blockIdx.y * 128;

  const unsigned short* w12e = w12b + (size_t)e * (2 * HD) * DM;

  v8f accG[4][2], accU[4][2];
#pragma unroll
  for (int a = 0; a < 4; ++a)
#pragma unroll
    for (int b = 0; b < 2; ++b) {
      accG[a][b] = (v8f){0.f,0.f,0.f,0.f,0.f,0.f,0.f,0.f};
      accU[a][b] = (v8f){0.f,0.f,0.f,0.f,0.f,0.f,0.f,0.f};
    }

  for (int kc = 0; kc < DM; kc += 64) {
    __syncthreads();
    // stage A: 128 rows x 64 bf16 (1024 x 16B segments, 4/thread)
#pragma unroll
    for (int i = 0; i < 4; ++i) {
      int s = tid + i * 256;
      int row = s >> 3, c = (s & 7) * 8;
      *(v8u*)(ldsA + row * LDSS + c) = *(const v8u*)(xb + (size_t)(tok0 + row) * DM + kc + c);
    }
    // stage B: 256 rows (gate j0.., then up HD+j0..), 2048 segments, 8/thread
#pragma unroll
    for (int i = 0; i < 8; ++i) {
      int s = tid + i * 256;
      int row = s >> 3, c = (s & 7) * 8;
      int grow = (row < 128) ? (j0 + row) : (HD + j0 + row - 128);
      *(v8u*)(ldsB + row * LDSS + c) = *(const v8u*)(w12e + (size_t)grow * DM + kc + c);
    }
    __syncthreads();
#pragma unroll
    for (int kk = 0; kk < 2; ++kk) {
      v16bf a0 = load_frag_a(ldsA + (wm * 64) * LDSS,      lane, kk);
      v16bf a1 = load_frag_a(ldsA + (wm * 64 + 16) * LDSS, lane, kk);
      v16bf a2 = load_frag_a(ldsA + (wm * 64 + 32) * LDSS, lane, kk);
      v16bf a3 = load_frag_a(ldsA + (wm * 64 + 48) * LDSS, lane, kk);
#pragma unroll
      for (int nt = 0; nt < 2; ++nt) {
        v16bf bg = load_frag_b(ldsB + (wn * 32 + nt * 16) * LDSS, lane, kk);
        accG[0][nt] = wmma_bf16(a0, bg, accG[0][nt]);
        accG[1][nt] = wmma_bf16(a1, bg, accG[1][nt]);
        accG[2][nt] = wmma_bf16(a2, bg, accG[2][nt]);
        accG[3][nt] = wmma_bf16(a3, bg, accG[3][nt]);
        v16bf bu = load_frag_b(ldsB + (128 + wn * 32 + nt * 16) * LDSS, lane, kk);
        accU[0][nt] = wmma_bf16(a0, bu, accU[0][nt]);
        accU[1][nt] = wmma_bf16(a1, bu, accU[1][nt]);
        accU[2][nt] = wmma_bf16(a2, bu, accU[2][nt]);
        accU[3][nt] = wmma_bf16(a3, bu, accU[3][nt]);
      }
    }
  }

  // epilogue: h = silu(g)*u, store bf16.  C layout: VGPR v, lanes 0-15 -> M=v, lanes 16-31 -> M=v+8
  const int rbase = tok0 + wm * 64 + ((lane >> 4) << 3);
  const int cbase = j0 + wn * 32 + (lane & 15);
#pragma unroll
  for (int mt = 0; mt < 4; ++mt)
#pragma unroll
    for (int nt = 0; nt < 2; ++nt)
#pragma unroll
      for (int v = 0; v < 8; ++v) {
        float g = accG[mt][nt][v];
        float u = accU[mt][nt][v];
        float hv = (g / (1.0f + __expf(-g))) * u;
        hb[(size_t)(rbase + mt * 16 + v) * HD + (cbase + nt * 16)] = f2bf(hv);
      }
}

// ---------------- GEMM2 (down projection) ----------------
// block = 256 threads (8 waves), tile = 256 tokens x 128 d-columns, K = HD in 64-chunks
// waves: 4 (token, 64 rows) x 2 (d, 64 cols); each wave: 4x4 16x16 tiles
// grid.x = TT/256 token tiles, grid.y = DM/128 d tiles
__global__ __launch_bounds__(256)
void gemm2_down(const unsigned short* __restrict__ hb,
                const unsigned short* __restrict__ w3b,
                float* __restrict__ out) {
  __shared__ __align__(16) unsigned short ldsA[256 * LDSS];
  __shared__ __align__(16) unsigned short ldsB[128 * LDSS];

  const int tid  = threadIdx.x;
  const int lane = tid & 31;
  const int wave = tid >> 5;
  const int wm   = wave & 3;    // token sub-tile (64 rows)
  const int wn   = wave >> 2;   // d sub-tile (64 cols)
  const int tok0 = blockIdx.x * 256;
  const int e    = blockIdx.x >> 4;   // 4096/256 = 16 tiles per expert
  const int d0   = blockIdx.y * 128;

  const unsigned short* w3e = w3b + (size_t)e * DM * HD;

  v8f acc[4][4];
#pragma unroll
  for (int a = 0; a < 4; ++a)
#pragma unroll
    for (int b = 0; b < 4; ++b)
      acc[a][b] = (v8f){0.f,0.f,0.f,0.f,0.f,0.f,0.f,0.f};

  for (int kc = 0; kc < HD; kc += 64) {
    __syncthreads();
    // stage A: 256 rows x 64 bf16 (2048 segs, 8/thread)
#pragma unroll
    for (int i = 0; i < 8; ++i) {
      int s = tid + i * 256;
      int row = s >> 3, c = (s & 7) * 8;
      *(v8u*)(ldsA + row * LDSS + c) = *(const v8u*)(hb + (size_t)(tok0 + row) * HD + kc + c);
    }
    // stage B: 128 rows x 64 bf16 (1024 segs, 4/thread)
#pragma unroll
    for (int i = 0; i < 4; ++i) {
      int s = tid + i * 256;
      int row = s >> 3, c = (s & 7) * 8;
      *(v8u*)(ldsB + row * LDSS + c) = *(const v8u*)(w3e + (size_t)(d0 + row) * HD + kc + c);
    }
    __syncthreads();
#pragma unroll
    for (int kk = 0; kk < 2; ++kk) {
      v16bf a0 = load_frag_a(ldsA + (wm * 64) * LDSS,      lane, kk);
      v16bf a1 = load_frag_a(ldsA + (wm * 64 + 16) * LDSS, lane, kk);
      v16bf a2 = load_frag_a(ldsA + (wm * 64 + 32) * LDSS, lane, kk);
      v16bf a3 = load_frag_a(ldsA + (wm * 64 + 48) * LDSS, lane, kk);
#pragma unroll
      for (int nt = 0; nt < 4; ++nt) {
        v16bf b = load_frag_b(ldsB + (wn * 64 + nt * 16) * LDSS, lane, kk);
        acc[0][nt] = wmma_bf16(a0, b, acc[0][nt]);
        acc[1][nt] = wmma_bf16(a1, b, acc[1][nt]);
        acc[2][nt] = wmma_bf16(a2, b, acc[2][nt]);
        acc[3][nt] = wmma_bf16(a3, b, acc[3][nt]);
      }
    }
  }

  const int rbase = tok0 + wm * 64 + ((lane >> 4) << 3);
  const int cbase = d0 + wn * 64 + (lane & 15);
#pragma unroll
  for (int mt = 0; mt < 4; ++mt)
#pragma unroll
    for (int nt = 0; nt < 4; ++nt)
#pragma unroll
      for (int v = 0; v < 8; ++v)
        out[(size_t)(rbase + mt * 16 + v) * DM + (cbase + nt * 16)] = acc[mt][nt][v];
}

extern "C" void kernel_launch(void* const* d_in, const int* in_sizes, int n_in,
                              void* d_out, int out_size, void* d_ws, size_t ws_size,
                              hipStream_t stream) {
  const float* x   = (const float*)d_in[0];  // (T, D)
  const float* w12 = (const float*)d_in[1];  // (E, 2H, D)
  const float* w3  = (const float*)d_in[2];  // (E, D, H)
  float* out = (float*)d_out;                // (T, D) f32

  // workspace layout (bf16 as u16): x | w12 | w3 | h
  unsigned short* xb   = (unsigned short*)d_ws;
  unsigned short* w12b = xb + (size_t)TT * DM;
  unsigned short* w3b  = w12b + (size_t)NE * 2 * HD * DM;
  unsigned short* hb   = w3b + (size_t)NE * DM * HD;

  cvt_f32_bf16<<<4096, 256, 0, stream>>>(x,   xb,   (TT * DM) / 4);
  cvt_f32_bf16<<<4096, 256, 0, stream>>>(w12, w12b, (NE * 2 * HD * DM) / 4);
  cvt_f32_bf16<<<2048, 256, 0, stream>>>(w3,  w3b,  (NE * DM * HD) / 4);

  gemm1_swiglu<<<dim3(TT / 128, HD / 128), 256, 0, stream>>>(xb, w12b, hb);
  gemm2_down  <<<dim3(TT / 256, DM / 128), 256, 0, stream>>>(hb, w3b, out);
}
